// VectorQuantizer_57638461112644
// MI455X (gfx1250) — matrix operand order repaired and checked by reading, not compiled, not run on hardware
//
#include <hip/hip_runtime.h>
#include <hip/hip_bf16.h>
#include <math.h>

#define N_ROWS   65536      // 32*2048
#define DIM      256
#define K_CODES  4096
#define BETA_F   0.25f
#define EPS_F    1e-12f

#define B_PITCH  264        // padded LDS row pitch in halves (conflict-free)
#define TILE_C   32         // codes staged per iteration (two 16-wide WMMA tiles)
#define N_ITER   (K_CODES / TILE_C)   // 128

typedef __attribute__((ext_vector_type(16))) _Float16 v16h;
typedef __attribute__((ext_vector_type(8)))  _Float16 v8h;
typedef __attribute__((ext_vector_type(4)))  _Float16 v4h;
typedef __attribute__((ext_vector_type(8)))  float    v8f;

static __device__ __forceinline__ v16h concat8(v8h lo, v8h hi) {
  return __builtin_shufflevector(lo, hi, 0,1,2,3,4,5,6,7,8,9,10,11,12,13,14,15);
}

// ---------------------------------------------------------------------------
// Kernel 1: L2-normalize each 256-elem row of z_e, write f16 copy for WMMA.
// One wave32 per row (8 elems/lane), 8 rows per 256-thread block.
// ---------------------------------------------------------------------------
__global__ __launch_bounds__(256)
void vq_normalize_f16(const float* __restrict__ z_e, _Float16* __restrict__ zf16) {
  const int wave = threadIdx.x >> 5;
  const int lane = threadIdx.x & 31;
  const int row  = blockIdx.x * 8 + wave;

  const float4* p = (const float4*)(z_e + (size_t)row * DIM + lane * 8);
  float4 a = p[0], b = p[1];
  float ss = a.x*a.x + a.y*a.y + a.z*a.z + a.w*a.w
           + b.x*b.x + b.y*b.y + b.z*b.z + b.w*b.w;
#pragma unroll
  for (int off = 16; off >= 1; off >>= 1) ss += __shfl_xor(ss, off, 32);

  const float inv = 1.0f / fmaxf(sqrtf(ss), EPS_F);
  v8h o;
  o[0]=(_Float16)(a.x*inv); o[1]=(_Float16)(a.y*inv);
  o[2]=(_Float16)(a.z*inv); o[3]=(_Float16)(a.w*inv);
  o[4]=(_Float16)(b.x*inv); o[5]=(_Float16)(b.y*inv);
  o[6]=(_Float16)(b.z*inv); o[7]=(_Float16)(b.w*inv);
  *(v8h*)(zf16 + (size_t)row * DIM + lane * 8) = o;
}

// ---------------------------------------------------------------------------
// Kernel 2: codebook f32 -> f16 (already normalized by setup).
// ---------------------------------------------------------------------------
__global__ __launch_bounds__(256)
void vq_emb_to_f16(const float* __restrict__ emb, _Float16* __restrict__ ef16) {
  const int i = blockIdx.x * blockDim.x + threadIdx.x;   // 4 elems per thread
  float4 v = ((const float4*)emb)[i];
  v4h o;
  o[0]=(_Float16)v.x; o[1]=(_Float16)v.y; o[2]=(_Float16)v.z; o[3]=(_Float16)v.w;
  ((v4h*)ef16)[i] = o;
}

// ---------------------------------------------------------------------------
// Kernel 3: zero counts + loss accumulator scratch.
// ---------------------------------------------------------------------------
__global__ void vq_zero(float* __restrict__ p, int n) {
  int i = blockIdx.x * blockDim.x + threadIdx.x;
  if (i < n) p[i] = 0.0f;
}

// ---------------------------------------------------------------------------
// Kernel 4: GEMM + argmax. Per block: 8 waves x 16 rows = 128 rows.
// A (16x256 f16) pinned in registers (8 x v16h). Codebook streamed 32 codes
// per iteration through double-buffered padded LDS; two independent WMMA
// accumulator chains (cols 0-15 and 16-31) interleave to hide WMMA RAW
// hazards and ds_load latency. Next tile's global loads are issued before the
// WMMA block and ds_stored into the alternate buffer after it, so global
// latency overlaps 16 WMMAs; one s_barrier per iteration.
//
// C/D layout: VGPR g of lane l holds row M=g+8*(l>>4), col N=l&15  (ISA 7.12.2)
// A layout  : lane l<16 row M=l, halves = K{0..7,16..23}; lane>=16 K{8..15,24..31}
// B layout  : lane l<16 col N=l, halves = K{0..7,16..23}; lane>=16 K{8..15,24..31}
// ---------------------------------------------------------------------------
__global__ __launch_bounds__(256)
void vq_argmax_wmma(const _Float16* __restrict__ zf16,
                    const _Float16* __restrict__ ef16,
                    int* __restrict__ idxOut) {
  __shared__ _Float16 sB[2 * TILE_C * B_PITCH];   // 2 x 32 x 264 halves = 33.8 KB

  const int tid    = threadIdx.x;
  const int wave   = tid >> 5;
  const int lane   = tid & 31;
  const int col    = lane & 15;
  const int hiHalf = lane >> 4;
  const int rowBase = blockIdx.x * 128 + wave * 16;

  // Load this wave's A tile into registers (reused across all 128 iterations).
  v16h areg[8];
  const _Float16* aRow = zf16 + (size_t)(rowBase + col) * DIM;
  const int aOff = hiHalf ? 8 : 0;
#pragma unroll
  for (int c = 0; c < 8; ++c) {
    const v8h* p = (const v8h*)(aRow + c * 32 + aOff);
    areg[c] = concat8(p[0], p[2]);           // +16 halves for the second group
  }

  float bestV[8]; int bestI[8];
#pragma unroll
  for (int g = 0; g < 8; ++g) { bestV[g] = -3.0e38f; bestI[g] = 0; }

  // Cooperative stage mapping: 32 rows x 512 B; 64 B (4 x uint4) per thread.
  const int ldRow  = tid >> 3;               // 0..31 : code row within tile
  const int ldColH = (tid & 7) * 32;         // halves (64 B chunks)
  const int bOff   = hiHalf ? 8 : 0;

  // Prologue: stage tile 0 into buffer 0.
  {
    const uint4* s = (const uint4*)(ef16 + (size_t)ldRow * DIM + ldColH);
    uint4* d = (uint4*)(sB + ldRow * B_PITCH + ldColH);
    d[0] = s[0]; d[1] = s[1]; d[2] = s[2]; d[3] = s[3];
  }
  __syncthreads();

  for (int t = 0; t < N_ITER; ++t) {
    // Issue next tile's global loads now; latency hides under the WMMAs.
    uint4 r0, r1, r2, r3;
    if (t + 1 < N_ITER) {
      const uint4* s = (const uint4*)(ef16 +
          ((size_t)(t + 1) * TILE_C + ldRow) * DIM + ldColH);
      r0 = s[0]; r1 = s[1]; r2 = s[2]; r3 = s[3];
      __builtin_prefetch(ef16 + ((size_t)(t + 2) * TILE_C + ldRow) * DIM + ldColH,
                         0, 1);              // global_prefetch_b8 for t+2
    }

    const _Float16* sCur = sB + (t & 1) * (TILE_C * B_PITCH);
    const _Float16* b0 = sCur + col * B_PITCH + bOff;          // codes  0..15
    const _Float16* b1 = sCur + (col + 16) * B_PITCH + bOff;   // codes 16..31

    v8f acc0 = {}, acc1 = {};
#pragma unroll
    for (int c = 0; c < 8; ++c) {
      const v8h* p0 = (const v8h*)(b0 + c * 32);
      const v8h* p1 = (const v8h*)(b1 + c * 32);
      v16h vb0 = concat8(p0[0], p0[2]);
      v16h vb1 = concat8(p1[0], p1[2]);
      acc0 = __builtin_amdgcn_wmma_f32_16x16x32_f16(
               false, areg[c], false, vb0, (short)0, acc0, false, false);
      acc1 = __builtin_amdgcn_wmma_f32_16x16x32_f16(
               false, areg[c], false, vb1, (short)0, acc1, false, false);
    }

    // Running argmax; strict '>' keeps the earliest (lowest) code on ties,
    // and acc0 (lower codes) is folded in before acc1.
    const int code0 = t * TILE_C + col;
#pragma unroll
    for (int g = 0; g < 8; ++g)
      if (acc0[g] > bestV[g]) { bestV[g] = acc0[g]; bestI[g] = code0; }
#pragma unroll
    for (int g = 0; g < 8; ++g)
      if (acc1[g] > bestV[g]) { bestV[g] = acc1[g]; bestI[g] = code0 + 16; }

    // Stage next tile into the alternate buffer (safe: that buffer's readers
    // all passed the previous barrier).
    if (t + 1 < N_ITER) {
      uint4* d = (uint4*)(sB + ((t + 1) & 1) * (TILE_C * B_PITCH) +
                          ldRow * B_PITCH + ldColH);
      d[0] = r0; d[1] = r1; d[2] = r2; d[3] = r3;
    }
    __syncthreads();
  }

  // Reduce the 16 column-lanes of each row; prefer lower index on ties
  // (matches argmin-first-occurrence of the reference).
#pragma unroll
  for (int g = 0; g < 8; ++g) {
    float v = bestV[g]; int i = bestI[g];
#pragma unroll
    for (int off = 8; off >= 1; off >>= 1) {
      float ov = __shfl_xor(v, off, 32);
      int   oi = __shfl_xor(i, off, 32);
      if (ov > v || (ov == v && oi < i)) { v = ov; i = oi; }
    }
    if (col == 0) {
      const int row = rowBase + (hiHalf << 3) + g;
      idxOut[row] = i;
    }
  }
}

// ---------------------------------------------------------------------------
// Kernel 5: gather z_q = emb[idx] (f32), write it as z_q_st (value-identical),
// accumulate sum((z_q - z_e_norm)^2), histogram counts, indices as float.
// One wave per row; one atomicAdd per block for the loss partial.
// ---------------------------------------------------------------------------
__global__ __launch_bounds__(256)
void vq_gather_loss(const float* __restrict__ z_e,
                    const float* __restrict__ emb,
                    const int*   __restrict__ idx,
                    float* __restrict__ outQ,
                    float* __restrict__ outIdxF,
                    float* __restrict__ counts,
                    float* __restrict__ lossAccum) {
  __shared__ float part[8];
  const int wave = threadIdx.x >> 5;
  const int lane = threadIdx.x & 31;
  const int row  = blockIdx.x * 8 + wave;
  const int code = idx[row];

  const float4* e = (const float4*)(emb + (size_t)code * DIM + lane * 8);
  float4 q0 = e[0], q1 = e[1];
  const float4* z = (const float4*)(z_e + (size_t)row * DIM + lane * 8);
  float4 z0 = z[0], z1 = z[1];

  float ss = z0.x*z0.x + z0.y*z0.y + z0.z*z0.z + z0.w*z0.w
           + z1.x*z1.x + z1.y*z1.y + z1.z*z1.z + z1.w*z1.w;
#pragma unroll
  for (int off = 16; off >= 1; off >>= 1) ss += __shfl_xor(ss, off, 32);
  const float inv = 1.0f / fmaxf(sqrtf(ss), EPS_F);

  float sq = 0.0f, d;
  d = q0.x - z0.x*inv; sq += d*d;   d = q0.y - z0.y*inv; sq += d*d;
  d = q0.z - z0.z*inv; sq += d*d;   d = q0.w - z0.w*inv; sq += d*d;
  d = q1.x - z1.x*inv; sq += d*d;   d = q1.y - z1.y*inv; sq += d*d;
  d = q1.z - z1.z*inv; sq += d*d;   d = q1.w - z1.w*inv; sq += d*d;

  float4* o = (float4*)(outQ + (size_t)row * DIM + lane * 8);
  o[0] = q0; o[1] = q1;

#pragma unroll
  for (int off = 16; off >= 1; off >>= 1) sq += __shfl_xor(sq, off, 32);
  if (lane == 0) {
    part[wave] = sq;
    atomicAdd(&counts[code], 1.0f);
    outIdxF[row] = (float)code;
  }
  __syncthreads();
  if (threadIdx.x == 0) {
    float s = 0.0f;
#pragma unroll
    for (int w = 0; w < 8; ++w) s += part[w];
    atomicAdd(lossAccum, s);
  }
}

// ---------------------------------------------------------------------------
// Kernel 6: losses + perplexity (single block).
// ---------------------------------------------------------------------------
__global__ __launch_bounds__(256)
void vq_finalize(const float* __restrict__ counts,
                 const float* __restrict__ lossAccum,
                 float* __restrict__ outScalars) {
  __shared__ float red[256];
  float s = 0.0f;
  for (int i = threadIdx.x; i < K_CODES; i += 256) {
    float p = counts[i] * (1.0f / (float)N_ROWS);
    s += p * logf(p + 1e-10f);
  }
  red[threadIdx.x] = s;
  __syncthreads();
  for (int st = 128; st > 0; st >>= 1) {
    if (threadIdx.x < st) red[threadIdx.x] += red[threadIdx.x + st];
    __syncthreads();
  }
  if (threadIdx.x == 0) {
    const float cb = lossAccum[0] / ((float)N_ROWS * (float)DIM);
    outScalars[0] = cb * (1.0f + BETA_F);   // vq_loss
    outScalars[1] = cb;                     // codebook_loss
    outScalars[2] = cb;                     // commitment_loss (value-identical)
    outScalars[3] = expf(-red[0]);          // perplexity
  }
}

// ---------------------------------------------------------------------------
extern "C" void kernel_launch(void* const* d_in, const int* in_sizes, int n_in,
                              void* d_out, int out_size, void* d_ws, size_t ws_size,
                              hipStream_t stream) {
  const float* z_e = (const float*)d_in[0];   // [32,2048,256] f32
  const float* emb = (const float*)d_in[1];   // [4096,256]    f32
  float* out = (float*)d_out;

  // workspace layout
  char* ws = (char*)d_ws;
  _Float16* zf16  = (_Float16*)ws;                                    // 32 MB
  _Float16* ef16  = (_Float16*)(ws + (size_t)N_ROWS * DIM * 2);       //  2 MB
  int*      idx   = (int*)((char*)ef16 + (size_t)K_CODES * DIM * 2);  // 256 KB
  float*    counts = (float*)((char*)idx + (size_t)N_ROWS * 4);       // 16 KB
  float*    lossA  = counts + K_CODES;                                // 4 B

  // output layout (flat, return order)
  const size_t scalarBase = (size_t)N_ROWS * DIM;  // 16777216
  float* outScalars = out + scalarBase;            // vq, cb, cm, ppl
  float* outIdxF    = out + scalarBase + 4;        // indices (as float)

  vq_zero<<<(K_CODES + 1 + 255) / 256, 256, 0, stream>>>(counts, K_CODES + 1);
  vq_normalize_f16<<<N_ROWS / 8, 256, 0, stream>>>(z_e, zf16);
  vq_emb_to_f16<<<(K_CODES * DIM / 4) / 256, 256, 0, stream>>>(emb, ef16);
  vq_argmax_wmma<<<N_ROWS / 128, 256, 0, stream>>>(zf16, ef16, idx);
  vq_gather_loss<<<N_ROWS / 8, 256, 0, stream>>>(z_e, emb, idx, out, outIdxF,
                                                 counts, lossA);
  vq_finalize<<<1, 256, 0, stream>>>(counts, lossA, outScalars);
}
